// Gemma2AttentionWithExpert_41893111005336
// MI455X (gfx1250) — matrix operand order repaired and checked by reading, not compiled, not run on hardware
//
#include <hip/hip_runtime.h>
#include <hip/hip_bf16.h>

// ---------------------------------------------------------------------------
// Gemma2-style two-stream GQA attention for MI455X (gfx1250, wave32, WMMA).
// Matrix math on v_wmma_f32_16x16x32_bf16; LDS staging via the CDNA5 async
// copy engine (global_load_async_to_lds_b128 / ASYNCcnt), double-buffered.
// ---------------------------------------------------------------------------

typedef __attribute__((ext_vector_type(16))) __bf16 bf16x16;
typedef __attribute__((ext_vector_type(8)))  float  f32x8;

#define B_    4
#define L1_   1024
#define L2_   512
#define L_    1536
#define DPG_  2304
#define DEX_  1024
#define H_    8
#define HKV_  4
#define DH_   256
#define SCALE_ 0.0625f   // 256^-0.5

union FragBF {
    bf16x16 v;
    uint4   q[2];
};

__device__ __forceinline__ unsigned short f2bf(float x) {
    unsigned int u = __float_as_uint(x);
    unsigned int r = u + 0x7FFFu + ((u >> 16) & 1u);   // round-to-nearest-even
    return (unsigned short)(r >> 16);
}

__device__ __forceinline__ f32x8 wmma_bf16(const FragBF& a, const FragBF& b, f32x8 c) {
    return __builtin_amdgcn_wmma_f32_16x16x32_bf16(
        /*neg_a=*/false, a.v, /*neg_b=*/false, b.v,
        /*c_mod=*/(short)0, c, /*reuse_a=*/false, /*reuse_b=*/false);
}

// Flat LDS aperture: low 32 bits of a generic __shared__ pointer are the
// wave-relative LDS byte address, i.e. exactly what the async op's VDST wants.
__device__ __forceinline__ unsigned lds_off32(const void* p) {
    return (unsigned)(unsigned long long)p;
}

// CDNA5 async copy: Memory -> LDS, 16B per lane, tracked by ASYNCcnt.
__device__ __forceinline__ void async_ld_b128(unsigned lds, const void* g) {
    asm volatile("global_load_async_to_lds_b128 %0, %1, off"
                 :: "v"(lds), "v"(g) : "memory");
}
__device__ __forceinline__ void wait_async0() {
    asm volatile("s_wait_asynccnt 0x0" ::: "memory");
}

// ---------------------------------------------------------------------------
// fp32 -> bf16 elementwise conversion
// ---------------------------------------------------------------------------
__global__ void cvt_f32_bf16(const float* __restrict__ x,
                             unsigned short* __restrict__ y, long n) {
    long i = ((long)blockIdx.x * blockDim.x + threadIdx.x) * 4;
    if (i + 3 < n) {
        float4 v = *(const float4*)(x + i);
        ushort4 o;
        o.x = f2bf(v.x); o.y = f2bf(v.y); o.z = f2bf(v.z); o.w = f2bf(v.w);
        *(ushort4*)(y + i) = o;
    } else {
        for (; i < n; ++i) y[i] = f2bf(x[i]);
    }
}

// ---------------------------------------------------------------------------
// RoPE (fp32 math) + bf16 emit.  X: [B, L, nh, DH] f32, Y: same in bf16.
// ---------------------------------------------------------------------------
__global__ void rope_cvt(const float* __restrict__ X,
                         const float* __restrict__ cosb,
                         const float* __restrict__ sinb,
                         unsigned short* __restrict__ Y, int nh) {
    long idx = (long)blockIdx.x * blockDim.x + threadIdx.x;
    long total = (long)B_ * L_ * nh * (DH_ / 2);
    if (idx >= total) return;
    int  i   = (int)(idx & 127);          // DH/2 = 128
    long rem = idx >> 7;
    int  hh  = (int)(rem % nh);
    long bl  = rem / nh;                  // b*L + l
    float c = cosb[bl * 128 + i];
    float s = sinb[bl * 128 + i];
    const float* xp = X + (bl * nh + hh) * DH_;
    float x1 = xp[i], x2 = xp[i + 128];
    unsigned short* yp = Y + (bl * nh + hh) * DH_;
    yp[i]       = f2bf(x1 * c - x2 * s);
    yp[i + 128] = f2bf(x2 * c + x1 * s);
}

// ---------------------------------------------------------------------------
// Generic bf16 TN GEMM: C[M,N](f32) = A[M,K](bf16) * B[N,K]^T(bf16)
//   arow = (m / a_rp) * a_rs + a_ro + (m % a_rp)
//   crow = (m / c_rp) * c_rs + c_ro + (m % c_rp)
// Block tile 128x128, K-step 32; 8 wave32 waves each own a 32x64 tile.
// LDS double-buffered; tiles arrive via global_load_async_to_lds_b128 and the
// loop runs with ONE barrier per K-step:
//   [wait_async; barrier; issue(next tile -> other buffer); compute(this)]
// The issue is safe because the barrier proves every wave finished reading the
// target buffer in the previous iteration.
// ---------------------------------------------------------------------------
__global__ __launch_bounds__(256) void gemm_bf16(
    const unsigned short* __restrict__ A,
    const unsigned short* __restrict__ Bw,
    float* __restrict__ C,
    int M, int N, int K,
    int a_rp, int a_rs, int a_ro,
    int c_rp, int c_rs, int c_ro) {
    __shared__ __align__(16) unsigned short sA[2][128][40];  // 2 x 10 KB
    __shared__ __align__(16) unsigned short sB[2][128][40];

    const int tid  = threadIdx.x;
    const int lane = tid & 31, wid = tid >> 5;
    const int lm   = lane & 15, lc = lane >> 4;
    const int wm   = wid & 3,   wn = wid >> 2;
    const int m0   = blockIdx.y * 128;
    const int n0   = blockIdx.x * 128;
    const unsigned BUFA = sizeof(unsigned short) * 128 * 40;   // 10240

    // Per-thread staging coords: chunk cid = tid + 256*i -> row = tid/4 + 64i,
    // element base e = (tid&3)*8 (constant across i).  Hoist row remap + bases.
    const int e  = (tid & 3) * 8;
    const int r0 = tid >> 2;
    const unsigned short* agp[2];
    const unsigned short* bgp[2];
    unsigned la[2], lb[2];
#pragma unroll
    for (int i = 0; i < 2; ++i) {
        int row = r0 + 64 * i;
        int ar  = m0 + row;
        long arow = (long)(ar / a_rp) * a_rs + a_ro + (ar % a_rp);
        agp[i] = A  + arow * K + e;
        bgp[i] = Bw + (long)(n0 + row) * K + e;
        la[i]  = lds_off32(&sA[0][row][e]);
        lb[i]  = lds_off32(&sB[0][row][e]);
    }

    f32x8 acc[2][4];
#pragma unroll
    for (int i = 0; i < 2; ++i)
#pragma unroll
        for (int j = 0; j < 4; ++j)
            acc[i][j] = (f32x8){0.f, 0.f, 0.f, 0.f, 0.f, 0.f, 0.f, 0.f};

    auto issue = [&](int k0, int p) {
#pragma unroll
        for (int i = 0; i < 2; ++i) {
            async_ld_b128(la[i] + p * BUFA, agp[i] + k0);
            async_ld_b128(lb[i] + p * BUFA, bgp[i] + k0);
            // Speculative prefetch two tiles ahead (dropped silently if OOB).
            __builtin_prefetch(agp[i] + k0 + 64, 0, 2);
            __builtin_prefetch(bgp[i] + k0 + 64, 0, 2);
        }
    };

    const int ntile = K >> 5;
    issue(0, 0);
    for (int it = 0; it < ntile; ++it) {
        const int p = it & 1;
        wait_async0();
        __syncthreads();
        if (it + 1 < ntile) issue((it + 1) << 5, p ^ 1);

        const unsigned short (*pA)[40] = sA[p];
        const unsigned short (*pB)[40] = sB[p];
        FragBF fa[2], fb[4];
#pragma unroll
        for (int mi = 0; mi < 2; ++mi) {
            const unsigned short* rp = &pA[wm * 32 + mi * 16 + lm][0];
            fa[mi].q[0] = *(const uint4*)(rp + lc * 8);
            fa[mi].q[1] = *(const uint4*)(rp + 16 + lc * 8);
        }
#pragma unroll
        for (int ni = 0; ni < 4; ++ni) {
            const unsigned short* rp = &pB[wn * 64 + ni * 16 + lm][0];
            fb[ni].q[0] = *(const uint4*)(rp + lc * 8);
            fb[ni].q[1] = *(const uint4*)(rp + 16 + lc * 8);
        }
#pragma unroll
        for (int mi = 0; mi < 2; ++mi)
#pragma unroll
            for (int ni = 0; ni < 4; ++ni)
                acc[mi][ni] = wmma_bf16(fa[mi], fb[ni], acc[mi][ni]);
    }

    // C layout: VGPR r -> row (r + 8*lc), column lm within the 16x16 tile.
#pragma unroll
    for (int mi = 0; mi < 2; ++mi) {
#pragma unroll
        for (int r = 0; r < 8; ++r) {
            int gr = m0 + wm * 32 + mi * 16 + r + 8 * lc;
            long crow = (long)(gr / c_rp) * c_rs + c_ro + (gr % c_rp);
            float* cp = C + crow * N + n0 + wn * 64 + lm;
#pragma unroll
            for (int ni = 0; ni < 4; ++ni)
                cp[ni * 16] = acc[mi][ni][r];
        }
    }
}

// ---------------------------------------------------------------------------
// Flash attention (online softmax), GQA rep=2.
//   Q: [B, L, H, DH] bf16   K/V: [B, L, HKV, DH] bf16
//   mask: [B, L, L] f32     O: [B, L, H, DH] bf16 (normalized)
// 128 threads (4 waves), 64 query rows per block, K-tile = 32.
// K tiles ride the async engine (double-buffered); V tiles are transposed
// through VGPRs into LDS (scatter) — that work overlaps the in-flight async
// copy.  One barrier per K-step, same pipeline as the GEMM.
// ---------------------------------------------------------------------------
__global__ __launch_bounds__(128) void flash_attn(
    const unsigned short* __restrict__ Qb,
    const unsigned short* __restrict__ Kb,
    const unsigned short* __restrict__ Vb,
    const float* __restrict__ mask,
    unsigned short* __restrict__ O) {
    __shared__ __align__(16) unsigned short sK[2][32][DH_ + 8];  // 2 x 16.5 KB
    __shared__ __align__(16) unsigned short sVt[2][DH_][40];     // 2 x 20 KB
    __shared__ __align__(16) unsigned short sP[4][16][40];       // 5 KB

    const int tid  = threadIdx.x;
    const int lane = tid & 31, wid = tid >> 5;
    const int lm   = lane & 15, lc = lane >> 4;
    const int b    = blockIdx.z;
    const int h    = blockIdx.y;
    const int q0   = blockIdx.x * 64;
    const int kvh  = h >> 1;             // GQA: rep = H/HKV = 2
    const int qw   = q0 + wid * 16;
    const int KVROW = HKV_ * DH_;        // 1024 elements per kv row
    const unsigned BUFK = sizeof(unsigned short) * 32 * (DH_ + 8);

    // Per-thread staging coords: chunk cid = tid + 128*i -> kv row tid/32+4i,
    // dh element base e0 = (tid&31)*8 (constant across i).
    const int r0 = tid >> 5;
    const int e0 = (tid & 31) * 8;
    const unsigned short* kbase =
        Kb + ((long)b * L_ + r0) * KVROW + kvh * DH_ + e0;
    const unsigned short* vbase =
        Vb + ((long)b * L_ + r0) * KVROW + kvh * DH_ + e0;
    const unsigned lK0 = lds_off32(&sK[0][r0][e0]);

    // Q fragments for the full DH=256 (8 fragments of 16x32), straight
    // from global in WMMA A-layout.
    FragBF fq[8];
    {
        const unsigned short* qp =
            Qb + (((long)(b * L_ + qw + lm)) * H_ + h) * DH_;
#pragma unroll
        for (int f = 0; f < 8; ++f) {
            fq[f].q[0] = *(const uint4*)(qp + 32 * f + lc * 8);
            fq[f].q[1] = *(const uint4*)(qp + 32 * f + 16 + lc * 8);
        }
    }

    f32x8 o[16];
#pragma unroll
    for (int t = 0; t < 16; ++t)
        o[t] = (f32x8){0.f, 0.f, 0.f, 0.f, 0.f, 0.f, 0.f, 0.f};
    float mrow[8], lrow[8];
#pragma unroll
    for (int r = 0; r < 8; ++r) { mrow[r] = -1e30f; lrow[r] = 0.f; }

    auto issue_tile = [&](int k0, int p) {
        const unsigned short* kp = kbase + (long)k0 * KVROW;
        const unsigned short* vp = vbase + (long)k0 * KVROW;
        unsigned short (*vt)[40] = sVt[p];
#pragma unroll
        for (int i = 0; i < 8; ++i) {
            async_ld_b128(lK0 + p * BUFK + i * 4 * (DH_ + 8) * 2,
                          kp + i * 4 * KVROW);
            uint4 vv = *(const uint4*)(vp + i * 4 * KVROW);
            const unsigned short* vs = (const unsigned short*)&vv;
#pragma unroll
            for (int j = 0; j < 8; ++j) vt[e0 + j][r0 + 4 * i] = vs[j];
        }
    };

    const int NT = L_ / 32;   // 48 tiles
    issue_tile(0, 0);
    for (int it = 0; it < NT; ++it) {
        const int p  = it & 1;
        const int k0 = it * 32;
        wait_async0();
        __syncthreads();
        if (it + 1 < NT) issue_tile(k0 + 32, p ^ 1);

        const unsigned short (*pK)[DH_ + 8] = sK[p];
        const unsigned short (*pV)[40]      = sVt[p];

        // S tiles: s0 -> k cols [k0,k0+16), s1 -> [k0+16,k0+32)
        f32x8 s0 = (f32x8){0.f, 0.f, 0.f, 0.f, 0.f, 0.f, 0.f, 0.f};
        f32x8 s1 = s0;
#pragma unroll
        for (int f = 0; f < 8; ++f) {
            FragBF kb0, kb1;
            kb0.q[0] = *(const uint4*)(&pK[lm][32 * f + lc * 8]);
            kb0.q[1] = *(const uint4*)(&pK[lm][32 * f + 16 + lc * 8]);
            kb1.q[0] = *(const uint4*)(&pK[16 + lm][32 * f + lc * 8]);
            kb1.q[1] = *(const uint4*)(&pK[16 + lm][32 * f + 16 + lc * 8]);
            s0 = wmma_bf16(fq[f], kb0, s0);
            s1 = wmma_bf16(fq[f], kb1, s1);
        }

        // Online softmax.  Element r sits at row (qw + r + 8*lc), column lm;
        // row-stats reduce across the 16-lane half with xor shuffles.
        float alpha[8];
#pragma unroll
        for (int r = 0; r < 8; ++r) {
            int qrow = qw + r + 8 * lc;
            const float* mp = mask + ((long)(b * L_) + qrow) * L_ + k0 + lm;
            float v0 = s0[r] * SCALE_ + mp[0];
            float v1 = s1[r] * SCALE_ + mp[16];
            float t = fmaxf(v0, v1);
            t = fmaxf(t, __shfl_xor(t, 1, 32));
            t = fmaxf(t, __shfl_xor(t, 2, 32));
            t = fmaxf(t, __shfl_xor(t, 4, 32));
            t = fmaxf(t, __shfl_xor(t, 8, 32));
            float mn = fmaxf(mrow[r], t);
            alpha[r] = __expf(mrow[r] - mn);
            mrow[r]  = mn;
            float e0x = __expf(v0 - mn);
            float e1x = __expf(v1 - mn);
            float rs = e0x + e1x;
            rs += __shfl_xor(rs, 1, 32);
            rs += __shfl_xor(rs, 2, 32);
            rs += __shfl_xor(rs, 4, 32);
            rs += __shfl_xor(rs, 8, 32);
            lrow[r] = lrow[r] * alpha[r] + rs;
            sP[wid][r + 8 * lc][lm]      = f2bf(e0x);
            sP[wid][r + 8 * lc][16 + lm] = f2bf(e1x);
        }
#pragma unroll
        for (int t = 0; t < 16; ++t)
#pragma unroll
            for (int r = 0; r < 8; ++r) o[t][r] *= alpha[r];

        // Re-materialize P as an A-fragment (wave-private LDS; DS in-order).
        FragBF pa;
        pa.q[0] = *(const uint4*)(&sP[wid][lm][lc * 8]);
        pa.q[1] = *(const uint4*)(&sP[wid][lm][16 + lc * 8]);

        // O += P * V, one 16x16 tile per dh slice.
#pragma unroll
        for (int t = 0; t < 16; ++t) {
            FragBF vb;
            vb.q[0] = *(const uint4*)(&pV[16 * t + lm][lc * 8]);
            vb.q[1] = *(const uint4*)(&pV[16 * t + lm][16 + lc * 8]);
            o[t] = wmma_bf16(pa, vb, o[t]);
        }
    }

    // Normalize and emit bf16 att output [B, L, H*DH].
    float inv[8];
#pragma unroll
    for (int r = 0; r < 8; ++r) inv[r] = 1.f / lrow[r];
#pragma unroll
    for (int t = 0; t < 16; ++t) {
#pragma unroll
        for (int r = 0; r < 8; ++r) {
            int qrow = qw + r + 8 * lc;
            O[(((long)(b * L_ + qrow)) * H_ + h) * DH_ + 16 * t + lm] =
                f2bf(o[t][r] * inv[r]);
        }
    }
}

// ---------------------------------------------------------------------------
// Host-side orchestration.
// ---------------------------------------------------------------------------
extern "C" void kernel_launch(void* const* d_in, const int* in_sizes, int n_in,
                              void* d_out, int out_size, void* d_ws, size_t ws_size,
                              hipStream_t stream) {
    (void)in_sizes; (void)n_in; (void)out_size; (void)ws_size;

    const float* pg   = (const float*)d_in[0];
    const float* ex   = (const float*)d_in[1];
    const float* cosb = (const float*)d_in[2];
    const float* sinb = (const float*)d_in[3];
    const float* mask = (const float*)d_in[4];
    const float* Wq0  = (const float*)d_in[5];
    const float* Wk0  = (const float*)d_in[6];
    const float* Wv0  = (const float*)d_in[7];
    const float* Wo0  = (const float*)d_in[8];
    const float* Wq1  = (const float*)d_in[9];
    const float* Wk1  = (const float*)d_in[10];
    const float* Wv1  = (const float*)d_in[11];
    const float* Wo1  = (const float*)d_in[12];
    float* out = (float*)d_out;

    // ---- workspace carve-up (~240 MB) ----
    char* base = (char*)d_ws;
    size_t off = 0;
    auto carve = [&](size_t bytes) -> char* {
        char* p = base + off;
        off = (off + bytes + 255) & ~(size_t)255;
        return p;
    };
    const long nPG  = (long)B_ * L1_ * DPG_;
    const long nEX  = (long)B_ * L2_ * DEX_;
    const long nQ   = (long)B_ * L_ * H_ * DH_;
    const long nKV  = (long)B_ * L_ * HKV_ * DH_;

    unsigned short* pg_bf  = (unsigned short*)carve(nPG * 2);
    unsigned short* ex_bf  = (unsigned short*)carve(nEX * 2);
    unsigned short* wq0_bf = (unsigned short*)carve((long)2048 * DPG_ * 2);
    unsigned short* wk0_bf = (unsigned short*)carve((long)1024 * DPG_ * 2);
    unsigned short* wv0_bf = (unsigned short*)carve((long)1024 * DPG_ * 2);
    unsigned short* wo0_bf = (unsigned short*)carve((long)DPG_ * 2048 * 2);
    unsigned short* wq1_bf = (unsigned short*)carve((long)2048 * DEX_ * 2);
    unsigned short* wk1_bf = (unsigned short*)carve((long)1024 * DEX_ * 2);
    unsigned short* wv1_bf = (unsigned short*)carve((long)1024 * DEX_ * 2);
    unsigned short* wo1_bf = (unsigned short*)carve((long)DEX_ * 2048 * 2);
    float* qf = (float*)carve(nQ * 4);
    float* kf = (float*)carve(nKV * 4);
    float* vf = (float*)carve(nKV * 4);
    unsigned short* qbf   = (unsigned short*)carve(nQ * 2);
    unsigned short* kbf   = (unsigned short*)carve(nKV * 2);
    unsigned short* vbf   = (unsigned short*)carve(nKV * 2);
    unsigned short* attbf = (unsigned short*)carve(nQ * 2);

    auto cvt = [&](const float* x, unsigned short* y, long n) {
        long blocks = (n / 4 + 255) / 256;
        cvt_f32_bf16<<<(unsigned)blocks, 256, 0, stream>>>(x, y, n);
    };
    cvt(pg,  pg_bf,  nPG);
    cvt(ex,  ex_bf,  nEX);
    cvt(Wq0, wq0_bf, (long)2048 * DPG_);
    cvt(Wk0, wk0_bf, (long)1024 * DPG_);
    cvt(Wv0, wv0_bf, (long)1024 * DPG_);
    cvt(Wo0, wo0_bf, (long)DPG_ * 2048);
    cvt(Wq1, wq1_bf, (long)2048 * DEX_);
    cvt(Wk1, wk1_bf, (long)1024 * DEX_);
    cvt(Wv1, wv1_bf, (long)1024 * DEX_);
    cvt(Wo1, wo1_bf, (long)DEX_ * 2048);

    auto gemm = [&](const unsigned short* A, const unsigned short* Bw, float* C,
                    int M, int N, int K,
                    int arp, int ars, int aro,
                    int crp, int crs, int cro) {
        dim3 grid(N / 128, M / 128);
        gemm_bf16<<<grid, 256, 0, stream>>>(A, Bw, C, M, N, K,
                                            arp, ars, aro, crp, crs, cro);
    };

    // ---- QKV projections (sequence concat fused via C row remap) ----
    gemm(pg_bf, wq0_bf, qf, B_ * L1_, 2048, DPG_, B_ * L1_, 0, 0, L1_, L_, 0);
    gemm(pg_bf, wk0_bf, kf, B_ * L1_, 1024, DPG_, B_ * L1_, 0, 0, L1_, L_, 0);
    gemm(pg_bf, wv0_bf, vf, B_ * L1_, 1024, DPG_, B_ * L1_, 0, 0, L1_, L_, 0);
    gemm(ex_bf, wq1_bf, qf, B_ * L2_, 2048, DEX_, B_ * L2_, 0, 0, L2_, L_, L1_);
    gemm(ex_bf, wk1_bf, kf, B_ * L2_, 1024, DEX_, B_ * L2_, 0, 0, L2_, L_, L1_);
    gemm(ex_bf, wv1_bf, vf, B_ * L2_, 1024, DEX_, B_ * L2_, 0, 0, L2_, L_, L1_);

    // ---- RoPE (Q, K) + bf16 V ----
    {
        long tq = (long)B_ * L_ * H_ * (DH_ / 2);
        rope_cvt<<<(unsigned)((tq + 255) / 256), 256, 0, stream>>>(qf, cosb, sinb, qbf, H_);
        long tk = (long)B_ * L_ * HKV_ * (DH_ / 2);
        rope_cvt<<<(unsigned)((tk + 255) / 256), 256, 0, stream>>>(kf, cosb, sinb, kbf, HKV_);
        cvt(vf, vbf, nKV);
    }

    // ---- flash attention ----
    {
        dim3 grid(L_ / 64, H_, B_);
        flash_attn<<<grid, 128, 0, stream>>>(qbf, kbf, vbf, mask, attbf);
    }

    // ---- output projections straight into d_out ----
    gemm(attbf, wo0_bf, out, B_ * L1_, DPG_, 2048,
         L1_, L_, 0, B_ * L1_, 0, 0);
    gemm(attbf, wo1_bf, out + (long)B_ * L1_ * DPG_, B_ * L2_, DEX_, 2048,
         L2_, L_, L1_, B_ * L2_, 0, 0);
}
